// GCN_6227702579508
// MI455X (gfx1250) — matrix operand order repaired and checked by reading, not compiled, not run on hardware
//
#include <hip/hip_runtime.h>
#include <hip/hip_bf16.h>
#include <math.h>
#include <stdint.h>

#define N_NODES 50000
#define N_EDGES 800000
#define HDIM    128
#define FFDIM   512
#define K1DIM   384           // 3*H concat
#define SCALE_F 30.0f
#define LN_EPS  1e-5f
#define SCAN_CH 196           // ceil(N_NODES / 256)

typedef _Float16 half_t;
typedef __attribute__((ext_vector_type(16))) _Float16 v16h;
typedef __attribute__((ext_vector_type(8)))  _Float16 v8h;
typedef __attribute__((ext_vector_type(4)))  _Float16 v4h;
typedef __attribute__((ext_vector_type(8)))  float    v8f;
typedef __attribute__((ext_vector_type(4)))  float    v4f;

union VH { v16h v; v8h h[2]; };

// D = A(16x32 f16) x B(32x16 f16) + C(16x16 f32)
__device__ __forceinline__ v8f wmma16x16x32(v16h a, v16h b, v8f c) {
  return __builtin_amdgcn_wmma_f32_16x16x32_f16(false, a, false, b, (short)0, c,
                                                false, false);
}

__device__ __forceinline__ float gelu_exact(float x) {
  return 0.5f * x * (1.0f + erff(x * 0.7071067811865476f));
}

__device__ __forceinline__ float wave_allreduce_sum(float v) {
  for (int m = 16; m >= 1; m >>= 1) v += __shfl_xor(v, m, 32);
  return v;
}

// A-fragment (16x32 f16) from LDS, row-major [row][k] with row stride ldk.
// lanes 0-15: row=lane, halves 0-7 -> K 0-7, 8-15 -> K 16-23;
// lanes 16-31: row=lane-16, halves 0-7 -> K 8-15, 8-15 -> K 24-31.
__device__ __forceinline__ v16h load_afrag_lds(const half_t* base, int rowBase,
                                               int ldk, int kt, int lane) {
  int row = rowBase + (lane & 15);
  int k0  = kt * 32 + ((lane & 16) ? 8 : 0);
  const half_t* p = base + row * ldk + k0;
  VH u;
  u.h[0] = *(const v8h*)(p);        // ds_load_b128
  u.h[1] = *(const v8h*)(p + 16);   // ds_load_b128
  return u.v;
}

// B-fragment (32x16 f16) pre-packed in global ws: tile (kt,nt) = 512 halves,
// lane's 16 halves contiguous (one 32B load).
__device__ __forceinline__ v16h load_bfrag(const half_t* wp, int kt, int nt,
                                           int NT, int lane) {
  return *(const v16h*)(wp + (((kt * NT + nt) << 9) + (lane << 4)));
}

// 32-row MLP layer: out[32 x 128] = act(in[32 x (KT*32)] @ B + bias)
// M register-blocked x2 so each B-fragment is reused twice.
template <int KT, int LD, bool GELU>
__device__ __forceinline__ void mlp_layer32(const half_t* in, const half_t* W,
                                            const float* __restrict__ bias,
                                            half_t* out, int lane) {
  v8f C0[8], C1[8];
  for (int nt = 0; nt < 8; ++nt) { C0[nt] = {}; C1[nt] = {}; }
#pragma unroll
  for (int kt = 0; kt < KT; ++kt) {
    v16h a0 = load_afrag_lds(in, 0, LD, kt, lane);
    v16h a1 = load_afrag_lds(in, 16, LD, kt, lane);
#pragma unroll
    for (int nt = 0; nt < 8; ++nt) {
      v16h b = load_bfrag(W, kt, nt, 8, lane);
      C0[nt] = wmma16x16x32(a0, b, C0[nt]);
      C1[nt] = wmma16x16x32(a1, b, C1[nt]);
    }
  }
  const int nlo = lane & 15, mhi = (lane & 16) ? 8 : 0;
#pragma unroll
  for (int nt = 0; nt < 8; ++nt) {
    float bv = bias[nt * 16 + nlo];
#pragma unroll
    for (int r = 0; r < 8; ++r) {
      float v0 = C0[nt][r] + bv;
      float v1 = C1[nt][r] + bv;
      if (GELU) { v0 = gelu_exact(v0); v1 = gelu_exact(v1); }
      out[(r + mhi) * HDIM + nt * 16 + nlo]        = (half_t)v0;
      out[(16 + r + mhi) * HDIM + nt * 16 + nlo]   = (half_t)v1;
    }
  }
}

// ---------------------------------------------------------------------------
// prep: zero degree histogram + pack weights into f16 B-fragments.
// ---------------------------------------------------------------------------
__global__ void gcn_prep_kernel(const float* W1, const float* W2, const float* W3,
                                const float* Win, const float* Wout,
                                int* deg,
                                half_t* W1h, half_t* W2h, half_t* W3h,
                                half_t* Winh, half_t* Wouth) {
  long long tid = (long long)blockIdx.x * blockDim.x + threadIdx.x;
  if (tid < N_NODES) { deg[tid] = 0; return; }

  long long p = tid - N_NODES;
  const long long S1 = 12 * 8 * 512;   // W1: K=384, N=128
  const long long S2 = 4 * 8 * 512;    // W2: K=128, N=128
  const long long S3 = 4 * 8 * 512;    // W3
  const long long S4 = 4 * 32 * 512;   // W_in: K=128, N=512
  const long long S5 = 16 * 8 * 512;   // W_out: K=512, N=128
  const float* src; half_t* dst; int NT, ld;
  if      (p < S1)             { src = W1;   dst = W1h;   NT = 8;  ld = K1DIM; }
  else if (p < S1+S2)          { p -= S1;          src = W2;   dst = W2h;   NT = 8;  ld = HDIM; }
  else if (p < S1+S2+S3)       { p -= S1+S2;       src = W3;   dst = W3h;   NT = 8;  ld = HDIM; }
  else if (p < S1+S2+S3+S4)    { p -= S1+S2+S3;    src = Win;  dst = Winh;  NT = 32; ld = HDIM; }
  else if (p < S1+S2+S3+S4+S5) { p -= S1+S2+S3+S4; src = Wout; dst = Wouth; NT = 8;  ld = FFDIM; }
  else return;

  int tile = (int)(p >> 9), elem = (int)(p & 511);
  int lane = elem >> 4, hh = elem & 15;
  int kt = tile / NT, nt = tile - kt * NT;
  int n = nt * 16 + (lane & 15);
  int k = kt * 32 + ((lane & 16) ? 8 : 0) + ((hh < 8) ? hh : 8 + hh);
  dst[p] = (half_t)src[(long long)n * ld + k];
}

// --------------------------- CSR construction ------------------------------
__global__ void gcn_degree_kernel(const int* __restrict__ src_idx, int* deg) {
  int e = blockIdx.x * 256 + threadIdx.x;
  if (e < N_EDGES) atomicAdd(&deg[src_idx[e]], 1);
}

// Thread-coarsened single-block exclusive scan: each thread serially owns
// SCAN_CH contiguous nodes; one 8-step block scan over 256 partials
// (~16 barriers total instead of ~3000).
__global__ void __launch_bounds__(256)
gcn_scan_kernel(const int* __restrict__ deg, int* rowStart, int* cursor) {
  __shared__ int part[256];
  const int tid = threadIdx.x;
  const int i0  = tid * SCAN_CH;

  int s = 0;
  for (int j = 0; j < SCAN_CH; ++j) {
    int i = i0 + j;
    s += (i < N_NODES) ? deg[i] : 0;
  }
  part[tid] = s;
  __syncthreads();

  int acc = s;
  for (int off = 1; off < 256; off <<= 1) {
    int t = (tid >= off) ? part[tid - off] : 0;
    __syncthreads();
    acc += t;
    part[tid] = acc;
    __syncthreads();
  }
  int run = acc - s;   // exclusive prefix of this thread's chunk

  for (int j = 0; j < SCAN_CH; ++j) {
    int i = i0 + j;
    if (i < N_NODES) {
      int d = deg[i];
      rowStart[i] = run;
      cursor[i]   = run;
      run += d;
    }
  }
  if (tid == 255) rowStart[N_NODES] = run;   // total edge count
}

__global__ void gcn_fill_kernel(const int* __restrict__ src_idx,
                                int* cursor, int* edgeList) {
  int e = blockIdx.x * 256 + threadIdx.x;
  if (e < N_EDGES) {
    int p = atomicAdd(&cursor[src_idx[e]], 1);
    edgeList[p] = e;
  }
}

// ---------------------------------------------------------------------------
// edge MLP: single-wave workgroup, 32 edges per block (N_EDGES % 32 == 0).
// gather -> L1(384->128) gelu -> L2 gelu -> L3 -> msg[e] (f16, via async
// LDS->global store).
// ---------------------------------------------------------------------------
__global__ void __launch_bounds__(32)
gcn_edge_mlp_kernel(const float* __restrict__ hV, const float* __restrict__ hE,
                    const int* __restrict__ src_idx, const int* __restrict__ dst_idx,
                    const float* __restrict__ b1, const float* __restrict__ b2,
                    const float* __restrict__ b3,
                    const half_t* __restrict__ W1h, const half_t* __restrict__ W2h,
                    const half_t* __restrict__ W3h,
                    half_t* __restrict__ msg) {
  __shared__ __align__(32) half_t ef[32 * K1DIM];  // 24KB gathered concat f16
  __shared__ __align__(32) half_t h1[32 * HDIM];   //  8KB (also layer-3 out)
  __shared__ __align__(32) half_t h2[32 * HDIM];   //  8KB
  __shared__ int ssrc[32];
  __shared__ int sdst[32];

  const int lane = threadIdx.x;
  const int e0   = blockIdx.x * 32;

  ssrc[lane] = src_idx[e0 + lane];
  sdst[lane] = dst_idx[e0 + lane];
  __syncthreads();   // single-wave WG: S_NOP, LDS is in-order per wave

  // gather: 32 rows x 48 chunks of 8 floats -> f16 LDS
  for (int item = lane; item < 32 * 48; item += 32) {
    int row = item / 48, chunk = item - row * 48;
    int seg = chunk >> 4, c8 = chunk & 15;
    const float* sp;
    if (seg == 0)      sp = hV + (long long)ssrc[row] * HDIM + c8 * 8;
    else if (seg == 1) sp = hE + (long long)(e0 + row) * HDIM + c8 * 8;
    else               sp = hV + (long long)sdst[row] * HDIM + c8 * 8;
    v4f f0 = *(const v4f*)sp;
    v4f f1 = *(const v4f*)(sp + 4);
    v8h o;
    o[0] = (half_t)f0[0]; o[1] = (half_t)f0[1]; o[2] = (half_t)f0[2]; o[3] = (half_t)f0[3];
    o[4] = (half_t)f1[0]; o[5] = (half_t)f1[1]; o[6] = (half_t)f1[2]; o[7] = (half_t)f1[3];
    *(v8h*)&ef[row * K1DIM + chunk * 8] = o;
  }
  __syncthreads();

  mlp_layer32<12, K1DIM, true >(ef, W1h, b1, h1, lane);   // 384 -> 128, gelu
  __syncthreads();
  mlp_layer32<4,  HDIM,  true >(h1, W2h, b2, h2, lane);   // 128 -> 128, gelu
  __syncthreads();
  mlp_layer32<4,  HDIM,  false>(h2, W3h, b3, h1, lane);   // 128 -> 128 message

  // coalesced message writeback: 32 rows x 256B, LDS -> global, async path.
  {
    asm volatile("s_wait_dscnt 0x0" ::: "memory");  // h1 writes visible to async engine
    half_t* gbase = msg + (long long)e0 * HDIM;
#pragma unroll
    for (int it = 0; it < 16; ++it) {
      int off = it * 256 + lane * 8;                 // halves (16B per lane)
      unsigned lds = (unsigned)(uintptr_t)(&h1[off]);
      unsigned long long ga = (unsigned long long)(uintptr_t)(gbase + off);
      asm volatile("global_store_async_from_lds_b128 %0, %1, off"
                   :: "v"(ga), "v"(lds) : "memory");
    }
    asm volatile("s_wait_asynccnt 0x0" ::: "memory");
  }
}

// ---------------------------------------------------------------------------
// node update: 2 waves / 64 threads, 32 nodes per block.
// CSR mean of messages -> residual -> LN1 -> FFN(128->512->128) -> LN2.
// ---------------------------------------------------------------------------
__global__ void __launch_bounds__(64)
gcn_node_update_kernel(const float* __restrict__ hV,
                       const int* __restrict__ rowStart,
                       const int* __restrict__ edgeList,
                       const half_t* __restrict__ msg,
                       const float* __restrict__ ln1_g, const float* __restrict__ ln1_b,
                       const float* __restrict__ ln2_g, const float* __restrict__ ln2_b,
                       const float* __restrict__ b_in, const float* __restrict__ b_out,
                       const half_t* __restrict__ Winh, const half_t* __restrict__ Wouth,
                       float* __restrict__ out) {
  __shared__ __align__(32) float  Xf[32 * HDIM];   // 16KB residual x (f32)
  __shared__ __align__(32) half_t Xn[32 * HDIM];   //  8KB normalized (f16)
  __shared__ __align__(32) half_t H1[32 * FFDIM];  // 32KB ffn hidden (f16)

  const int tid  = threadIdx.x;
  const int lane = tid & 31;
  const int wave = tid >> 5;
  const int n0   = blockIdx.x * 32;
  const int r0   = wave * 16;
  const int f0   = lane * 4;
  const int nlo  = lane & 15;
  const int mhi  = (lane & 16) ? 8 : 0;

  // --- phase A: CSR message mean, residual, LN1 ---
  for (int i = 0; i < 16; ++i) {
    int row = r0 + i, node = n0 + row;
    float x[4];
    int rs = 0, re = 0;
    if (node < N_NODES) { rs = rowStart[node]; re = rowStart[node + 1]; }
    float acc[4] = {0.f, 0.f, 0.f, 0.f};
    for (int t = rs; t < re; ++t) {          // uniform across the wave
      int e = edgeList[t];
      v4h m = *(const v4h*)(msg + (long long)e * HDIM + f0);
      acc[0] += (float)m[0]; acc[1] += (float)m[1];
      acc[2] += (float)m[2]; acc[3] += (float)m[3];
    }
    if (node < N_NODES) {
      const float* hvp = hV + (long long)node * HDIM + f0;
      float inv = 1.0f / (fmaxf((float)(re - rs), 1.0f) * SCALE_F);
      for (int j = 0; j < 4; ++j) x[j] = hvp[j] + acc[j] * inv;
    } else {
      for (int j = 0; j < 4; ++j) x[j] = 0.0f;
    }
    float mean = wave_allreduce_sum(x[0] + x[1] + x[2] + x[3]) * (1.0f / HDIM);
    float vs = 0.0f;
    for (int j = 0; j < 4; ++j) { float d = x[j] - mean; vs += d * d; }
    float rstd = rsqrtf(wave_allreduce_sum(vs) * (1.0f / HDIM) + LN_EPS);
    for (int j = 0; j < 4; ++j) {
      int f = f0 + j;
      Xf[row * HDIM + f] = x[j];
      Xn[row * HDIM + f] = (half_t)((x[j] - mean) * rstd * ln1_g[f] + ln1_b[f]);
    }
  }
  __syncthreads();

  // --- phase B: h1 = gelu(Xn @ W_in^T + b_in), A-frags cached in regs ---
  {
    v16h a4[4];
#pragma unroll
    for (int kt = 0; kt < 4; ++kt) a4[kt] = load_afrag_lds(Xn, r0, HDIM, kt, lane);
    for (int nt = 0; nt < 32; ++nt) {
      v8f c = {};
#pragma unroll
      for (int kt = 0; kt < 4; ++kt)
        c = wmma16x16x32(a4[kt], load_bfrag(Winh, kt, nt, 32, lane), c);
      float bias = b_in[nt * 16 + nlo];
#pragma unroll
      for (int r = 0; r < 8; ++r)
        H1[(r0 + r + mhi) * FFDIM + nt * 16 + nlo] =
            (half_t)gelu_exact(c[r] + bias);
    }
  }
  __syncthreads();

  // --- phase C: dh = H1 @ W_out^T + b_out; all 16 A-frags cached in regs ---
  {
    v16h aC[16];
#pragma unroll
    for (int kt = 0; kt < 16; ++kt) aC[kt] = load_afrag_lds(H1, r0, FFDIM, kt, lane);
#pragma unroll
    for (int nt = 0; nt < 8; ++nt) {
      v8f c = {};
#pragma unroll
      for (int kt = 0; kt < 16; ++kt)
        c = wmma16x16x32(aC[kt], load_bfrag(Wouth, kt, nt, 8, lane), c);
      float bias = b_out[nt * 16 + nlo];
#pragma unroll
      for (int r = 0; r < 8; ++r)
        Xf[(r0 + r + mhi) * HDIM + nt * 16 + nlo] += c[r] + bias;
    }
  }
  __syncthreads();

  // --- phase D: LN2, write output ---
  for (int i = 0; i < 16; ++i) {
    int row = r0 + i, node = n0 + row;
    float x[4];
    for (int j = 0; j < 4; ++j) x[j] = Xf[row * HDIM + f0 + j];
    float mean = wave_allreduce_sum(x[0] + x[1] + x[2] + x[3]) * (1.0f / HDIM);
    float vs = 0.0f;
    for (int j = 0; j < 4; ++j) { float d = x[j] - mean; vs += d * d; }
    float rstd = rsqrtf(wave_allreduce_sum(vs) * (1.0f / HDIM) + LN_EPS);
    if (node < N_NODES) {
      float* op = out + (long long)node * HDIM + f0;
      for (int j = 0; j < 4; ++j) {
        int f = f0 + j;
        op[j] = (x[j] - mean) * rstd * ln2_g[f] + ln2_b[f];
      }
    }
  }
}

// ---------------------------------------------------------------------------
extern "C" void kernel_launch(void* const* d_in, const int* in_sizes, int n_in,
                              void* d_out, int out_size, void* d_ws, size_t ws_size,
                              hipStream_t stream) {
  const float* hV   = (const float*)d_in[0];
  const float* hE   = (const float*)d_in[1];
  const int*   src  = (const int*)d_in[2];
  // d_in[3] = batch_id (unused by the math)
  const int*   dst  = (const int*)d_in[4];
  const float* W1   = (const float*)d_in[5];
  const float* b1   = (const float*)d_in[6];
  const float* W2   = (const float*)d_in[7];
  const float* b2   = (const float*)d_in[8];
  const float* W3   = (const float*)d_in[9];
  const float* b3   = (const float*)d_in[10];
  const float* ln1g = (const float*)d_in[11];
  const float* ln1b = (const float*)d_in[12];
  const float* ln2g = (const float*)d_in[13];
  const float* ln2b = (const float*)d_in[14];
  const float* Win  = (const float*)d_in[15];
  const float* bin  = (const float*)d_in[16];
  const float* Wout = (const float*)d_in[17];
  const float* bout = (const float*)d_in[18];
  float* out = (float*)d_out;

  // workspace carve-up
  char* ws = (char*)d_ws;
  int* deg      = (int*)ws;    ws += (size_t)N_NODES * sizeof(int);
  int* rowStart = (int*)ws;    ws += (size_t)(N_NODES + 1) * sizeof(int);
  int* cursor   = (int*)ws;    ws += (size_t)N_NODES * sizeof(int);
  ws = (char*)(((uintptr_t)ws + 31) & ~(uintptr_t)31);
  int* edgeList = (int*)ws;    ws += (size_t)N_EDGES * sizeof(int);
  half_t* msg   = (half_t*)ws; ws += (size_t)N_EDGES * HDIM * sizeof(half_t); // 204.8MB
  half_t* W1h   = (half_t*)ws; ws += (size_t)12 * 8 * 512 * sizeof(half_t);
  half_t* W2h   = (half_t*)ws; ws += (size_t)4 * 8 * 512 * sizeof(half_t);
  half_t* W3h   = (half_t*)ws; ws += (size_t)4 * 8 * 512 * sizeof(half_t);
  half_t* Winh  = (half_t*)ws; ws += (size_t)4 * 32 * 512 * sizeof(half_t);
  half_t* Wouth = (half_t*)ws; ws += (size_t)16 * 8 * 512 * sizeof(half_t);

  {
    long long total = (long long)N_NODES +
                      (long long)(12 * 8 + 4 * 8 + 4 * 8 + 4 * 32 + 16 * 8) * 512;
    int blocks = (int)((total + 255) / 256);
    gcn_prep_kernel<<<blocks, 256, 0, stream>>>(W1, W2, W3, Win, Wout, deg,
                                                W1h, W2h, W3h, Winh, Wouth);
  }
  gcn_degree_kernel<<<(N_EDGES + 255) / 256, 256, 0, stream>>>(src, deg);
  gcn_scan_kernel<<<1, 256, 0, stream>>>(deg, rowStart, cursor);
  gcn_fill_kernel<<<(N_EDGES + 255) / 256, 256, 0, stream>>>(src, cursor, edgeList);
  gcn_edge_mlp_kernel<<<N_EDGES / 32, 32, 0, stream>>>(
      hV, hE, src, dst, b1, b2, b3, W1h, W2h, W3h, msg);
  gcn_node_update_kernel<<<(N_NODES + 31) / 32, 64, 0, stream>>>(
      hV, rowStart, edgeList, msg, ln1g, ln1b, ln2g, ln2b, bin, bout,
      Winh, Wouth, out);
}
